// GINModel_83004537962844
// MI455X (gfx1250) — compile-verified, matrix operand order checked by reading, not run on hardware
//
#include <hip/hip_runtime.h>
#include <hip/hip_bf16.h>
#include <math.h>

// Problem constants (fixed by the reference)
#define NN   100000
#define EE   256000
#define GG   4000
#define HH   300
#define EDD  64
#define LL   5
#define H2   600
#define HP   320    // H padded to multiple of 32
#define H2P  608    // 2H padded to multiple of 32
#define HD2  150
#define BN_EPS 1e-5f

// GEMM tiling: block = 128 M x 64 N; K staged through LDS in 256-half chunks.
// LDS row stride = 264 halves (132 dwords): TDM pad fields (interval=128 DW,
// amount=4 DW) produce it, and 132 mod 64 = 4 spreads the 16 fragment rows
// across distinct 4-bank groups.
#define GEMM_KC    256
#define GEMM_LDSK  264

typedef __attribute__((ext_vector_type(16))) _Float16 v16h;
typedef __attribute__((ext_vector_type(8)))  _Float16 v8h;
typedef __attribute__((ext_vector_type(8)))  float    v8f;
typedef __attribute__((ext_vector_type(4)))  unsigned int v4u;
typedef __attribute__((ext_vector_type(8)))  int      v8i;
typedef __attribute__((ext_vector_type(4)))  int      v4i;
typedef __attribute__((address_space(3)))    _Float16 lds_f16;

__device__ __forceinline__ float silu_f(float x) { return x / (1.f + __expf(-x)); }

// ---------------------------------------------------------------- utilities
__global__ void k_zero(float* p, size_t n) {
    size_t i = (size_t)blockIdx.x * blockDim.x + threadIdx.x;
    size_t stride = (size_t)gridDim.x * blockDim.x;
    for (; i < n; i += stride) p[i] = 0.f;
}

__global__ void k_copy(float* dst, const float* src, size_t n) {
    size_t i = (size_t)blockIdx.x * blockDim.x + threadIdx.x;
    size_t stride = (size_t)gridDim.x * blockDim.x;
    for (; i < n; i += stride) dst[i] = src[i];
}

// out[r, :] = emb[idx[r], :]
__global__ void k_gather_rows(const float* __restrict__ emb, const int* __restrict__ idx,
                              float* __restrict__ out, int n, int d) {
    size_t i = (size_t)blockIdx.x * blockDim.x + threadIdx.x;
    if (i >= (size_t)n * d) return;
    int r = (int)(i / d), j = (int)(i % d);
    out[i] = emb[(size_t)idx[r] * d + j];
}

__global__ void k_gather_cvt(const float* __restrict__ emb, const int* __restrict__ idx,
                             _Float16* __restrict__ out, int n, int d) {
    size_t i = (size_t)blockIdx.x * blockDim.x + threadIdx.x;
    if (i >= (size_t)n * d) return;
    int r = (int)(i / d), j = (int)(i % d);
    out[i] = (_Float16)emb[(size_t)idx[r] * d + j];
}

// f32 [rows,k] -> f16 [rows,kp] zero-padded in K
__global__ void k_cvt_pad(const float* __restrict__ src, _Float16* __restrict__ dst,
                          int rows, int k, int kp) {
    size_t i = (size_t)blockIdx.x * blockDim.x + threadIdx.x;
    if (i >= (size_t)rows * kp) return;
    int r = (int)(i / kp), j = (int)(i % kp);
    dst[i] = (j < k) ? (_Float16)src[(size_t)r * k + j] : (_Float16)0.f;
}

// -------------------------------------------------- WMMA GEMM: C = A @ W^T (+bias) (+=)
// A: f16 [M, Kp] row-major; W: f16 [Nout, Kp] row-major (B column n == W row n).
// Block super-tile 128M x 64N, 8 waves each own 16M x 64N.
// Weight slab staged into LDS: full 256-half K chunks go through the Tensor Data
// Mover (async DMA, TENSORcnt, LDS-stride padding via descriptor); tail chunks
// are staged manually. Per K-step: 1 A frag (global) + 4 B frags (ds_load_b128)
// feed 4 independent WMMAs.
__global__ void __launch_bounds__(256)
k_wmma_gemm(const _Float16* __restrict__ A, const _Float16* __restrict__ W,
            const float* __restrict__ bias, float* __restrict__ C,
            int M, int Nout, int Kp, int accumulate) {
    __shared__ __align__(16) _Float16 sB[64 * GEMM_LDSK];   // 33,792 bytes
    const int lane = threadIdx.x & 31;
    const int wave = threadIdx.x >> 5;
    const int m0 = (blockIdx.y * 8 + wave) * 16;
    const int n0 = blockIdx.x * 64;
    // no early exit: all waves participate in staging barriers; out-of-range
    // waves compute on clamped rows and skip stores.

    // ISA 16-bit fragment layout: lanes 0-15 carry K 0..7 & 16..23,
    // lanes 16-31 carry K 8..15 & 24..31 (each VGPR packs 2 halves).
    const int khalf = (lane >> 4) * 8;
    int arow = m0 + (lane & 15); if (arow >= M) arow = M - 1;
    const _Float16* Ap = A + (size_t)arow * Kp + khalf;

    const unsigned ldsBase = (unsigned)(size_t)(lds_f16*)sB;   // LDS byte offset

    union Frag { v16h v; v8h h[2]; };
    v8f acc[4] = {};

    for (int k0 = 0; k0 < Kp; k0 += GEMM_KC) {
        const int kc = (Kp - k0 < GEMM_KC) ? (Kp - k0) : GEMM_KC;  // multiple of 32
        __syncthreads();   // previous chunk fully consumed
        if (kc == GEMM_KC) {
            // ---- TDM path: DMA 64 x 256-half tile, padded to 264-half LDS rows
            if (wave == 0) {
                unsigned long long ga =
                    (unsigned long long)(size_t)(W + (size_t)n0 * Kp + k0);
                unsigned td0 = (unsigned)(Kp - k0);     // remaining line length
                unsigned td1 = (unsigned)(Nout - n0);   // remaining rows (OOB -> 0)
                v4u g0;
                g0[0] = 1u;                                          // count=1
                g0[1] = ldsBase;                                     // lds_addr
                g0[2] = (unsigned)(ga & 0xFFFFFFFFull);              // global_addr lo
                g0[3] = (unsigned)((ga >> 32) & 0x1FFFFFFull) | (2u << 30);  // hi | type=2
                v8i g1;
                g1[0] = (1 << 16)      // data_size = 2 bytes
                      | (1 << 20)      // pad_enable
                      | (6 << 22)      // pad_interval: 128 DWORDs
                      | (3 << 25);     // pad_amount: 4 DWORDs
                g1[1] = (int)((td0 & 0xFFFFu) << 16);                 // tensor_dim0 lo
                g1[2] = (int)((td0 >> 16) | ((td1 & 0xFFFFu) << 16)); // dim0 hi | dim1 lo
                g1[3] = (int)((td1 >> 16) | ((unsigned)GEMM_KC << 16)); // dim1 hi | tile_dim0
                g1[4] = 64;                                           // tile_dim1=64, tile_dim2=0
                g1[5] = (int)(unsigned)Kp;                            // tensor_dim0_stride lo
                g1[6] = 0;
                g1[7] = 0;
                v4i gz4 = {0, 0, 0, 0};
                v8i gz8 = {0, 0, 0, 0, 0, 0, 0, 0};
                __builtin_amdgcn_tensor_load_to_lds(g0, g1, gz4, gz4, gz8, 0);
                __builtin_amdgcn_s_wait_tensorcnt(0);
            }
        } else {
            // ---- manual tail staging (kc = 64 or 96 halves)
            const int chunks = kc >> 3;   // v8h copies per row
            for (int idx = threadIdx.x; idx < 64 * chunks; idx += 256) {
                int r = idx / chunks;
                int c = (idx - r * chunks) << 3;
                int wrow = n0 + r; if (wrow >= Nout) wrow = Nout - 1;
                *(v8h*)(sB + r * GEMM_LDSK + c) =
                    *(const v8h*)(W + (size_t)wrow * Kp + k0 + c);
            }
        }
        __syncthreads();

        for (int k = 0; k < kc; k += 32) {
            Frag a;
            a.h[0] = *(const v8h*)(Ap + k0 + k);
            a.h[1] = *(const v8h*)(Ap + k0 + k + 16);
            if (k + 32 < kc) __builtin_prefetch(Ap + k0 + k + 32, 0, 3);

            const _Float16* sb = sB + (lane & 15) * GEMM_LDSK + khalf + k;
            Frag b0, b1, b2, b3;
            b0.h[0] = *(const v8h*)(sb);
            b0.h[1] = *(const v8h*)(sb + 16);
            b1.h[0] = *(const v8h*)(sb + 16 * GEMM_LDSK);
            b1.h[1] = *(const v8h*)(sb + 16 * GEMM_LDSK + 16);
            b2.h[0] = *(const v8h*)(sb + 32 * GEMM_LDSK);
            b2.h[1] = *(const v8h*)(sb + 32 * GEMM_LDSK + 16);
            b3.h[0] = *(const v8h*)(sb + 48 * GEMM_LDSK);
            b3.h[1] = *(const v8h*)(sb + 48 * GEMM_LDSK + 16);

            acc[0] = __builtin_amdgcn_wmma_f32_16x16x32_f16(
                         false, a.v, false, b0.v, (short)0, acc[0], false, false);
            acc[1] = __builtin_amdgcn_wmma_f32_16x16x32_f16(
                         false, a.v, false, b1.v, (short)0, acc[1], false, false);
            acc[2] = __builtin_amdgcn_wmma_f32_16x16x32_f16(
                         false, a.v, false, b2.v, (short)0, acc[2], false, false);
            acc[3] = __builtin_amdgcn_wmma_f32_16x16x32_f16(
                         false, a.v, false, b3.v, (short)0, acc[3], false, false);
        }
    }

    // C/D layout: lane n = lane&15; rows m = (lane>>4)*8 + vgpr
    const int mbase = m0 + (lane >> 4) * 8;
#pragma unroll
    for (int t = 0; t < 4; ++t) {
        const int cn = n0 + t * 16 + (lane & 15);
        if (cn >= Nout) continue;
        const float bv = bias ? bias[cn] : 0.f;
#pragma unroll
        for (int r = 0; r < 8; ++r) {
            int cm = mbase + r;
            if (cm < M) {
                size_t o = (size_t)cm * Nout + cn;
                float val = acc[t][r] + bv;
                if (accumulate) val += C[o];
                C[o] = val;
            }
        }
    }
}

// ------------------------------------------------ GINE message + scatter-add
__global__ void k_message_scatter(const float* __restrict__ h, const float* __restrict__ ep,
                                  const int* __restrict__ srcI, const int* __restrict__ dstI,
                                  float* __restrict__ agg) {
    const int CH = HH / 4;  // 75 float4 chunks per row
    size_t i = (size_t)blockIdx.x * blockDim.x + threadIdx.x;
    if (i >= (size_t)EE * CH) return;
    int e = (int)(i / CH), c = (int)(i % CH);
    int s = srcI[e], d = dstI[e];
    float4 hv = *(const float4*)(h + (size_t)s * HH + c * 4);
    float4 ev = *(const float4*)(ep + (size_t)e * HH + c * 4);
    float* out = agg + (size_t)d * HH + c * 4;
    atomicAdd(out + 0, fmaxf(hv.x + ev.x, 0.f));
    atomicAdd(out + 1, fmaxf(hv.y + ev.y, 0.f));
    atomicAdd(out + 2, fmaxf(hv.z + ev.z, 0.f));
    atomicAdd(out + 3, fmaxf(hv.w + ev.w, 0.f));
}

// z16 = f16pad((1+eps[l])*h + agg)
__global__ void k_z_cvt(const float* __restrict__ h, const float* __restrict__ agg,
                        const float* __restrict__ epsArr, int l, _Float16* __restrict__ z16) {
    size_t i = (size_t)blockIdx.x * blockDim.x + threadIdx.x;
    if (i >= (size_t)NN * HP) return;
    int r = (int)(i / HP), j = (int)(i % HP);
    if (j < HH) {
        size_t o = (size_t)r * HH + j;
        z16[i] = (_Float16)((1.f + epsArr[l]) * h[o] + agg[o]);
    } else {
        z16[i] = (_Float16)0.f;
    }
}

// out16 = f16pad(silu(batchnorm(in)))   bnp = [gamma|beta|mean|var] each length d
__global__ void k_bn_silu_cvt(const float* __restrict__ in, const float* __restrict__ bnp,
                              _Float16* __restrict__ out16, int rows, int d, int dp) {
    size_t i = (size_t)blockIdx.x * blockDim.x + threadIdx.x;
    if (i >= (size_t)rows * dp) return;
    int r = (int)(i / dp), j = (int)(i % dp);
    if (j < d) {
        float g = bnp[j], b = bnp[d + j], m = bnp[2 * d + j], v = bnp[3 * d + j];
        float x = (in[(size_t)r * d + j] - m) * (g * rsqrtf(v + BN_EPS)) + b;
        out16[i] = (_Float16)silu_f(x);
    } else {
        out16[i] = (_Float16)0.f;
    }
}

// h = silu(bn2(t2)) (+ h if add_res)
__global__ void k_bn2_residual(const float* __restrict__ t2, const float* __restrict__ bnp,
                               float* __restrict__ h, int rows, int d, int add_res) {
    size_t i = (size_t)blockIdx.x * blockDim.x + threadIdx.x;
    if (i >= (size_t)rows * d) return;
    int j = (int)(i % d);
    float g = bnp[j], b = bnp[d + j], m = bnp[2 * d + j], v = bnp[3 * d + j];
    float x = silu_f((t2[i] - m) * (g * rsqrtf(v + BN_EPS)) + b);
    if (add_res) x += h[i];
    h[i] = x;
}

// out[seg[r], :] += x[r, :]
__global__ void k_segment_sum(const float* __restrict__ x, const int* __restrict__ seg,
                              float* __restrict__ out, int n, int d) {
    const int CH = HH / 4;
    size_t i = (size_t)blockIdx.x * blockDim.x + threadIdx.x;
    if (i >= (size_t)n * CH) return;
    int r = (int)(i / CH), c = (int)(i % CH);
    int g = seg[r];
    float4 v = *(const float4*)(x + (size_t)r * d + c * 4);
    float* o = out + (size_t)g * d + c * 4;
    atomicAdd(o + 0, v.x);
    atomicAdd(o + 1, v.y);
    atomicAdd(o + 2, v.z);
    atomicAdd(o + 3, v.w);
}

// h[n,:] += vn[batch[n],:]
__global__ void k_add_broadcast(float* __restrict__ h, const float* __restrict__ vn,
                                const int* __restrict__ batch) {
    size_t i = (size_t)blockIdx.x * blockDim.x + threadIdx.x;
    if (i >= (size_t)NN * HH) return;
    int r = (int)(i / HH), j = (int)(i % HH);
    h[i] += vn[(size_t)batch[r] * HH + j];
}

// readout: LayerNorm -> SiLU -> Linear(H,150)+SiLU -> Linear(150,1); one block per graph
__global__ void __launch_bounds__(256)
k_readout(const float* __restrict__ hg, const float* __restrict__ lng,
          const float* __restrict__ lnb, const float* __restrict__ hW1,
          const float* __restrict__ hb1, const float* __restrict__ hW2,
          const float* __restrict__ hb2, float* __restrict__ out) {
    __shared__ float s_r[HH];
    __shared__ float s_y[HD2];
    __shared__ float s_red[256];
    const int g = blockIdx.x, t = threadIdx.x;

    float part = 0.f;
    for (int j = t; j < HH; j += 256) { float v = hg[(size_t)g * HH + j]; s_r[j] = v; part += v; }
    s_red[t] = part; __syncthreads();
    for (int s = 128; s > 0; s >>= 1) { if (t < s) s_red[t] += s_red[t + s]; __syncthreads(); }
    float mu = s_red[0] / HH; __syncthreads();

    part = 0.f;
    for (int j = t; j < HH; j += 256) { float d = s_r[j] - mu; part += d * d; }
    s_red[t] = part; __syncthreads();
    for (int s = 128; s > 0; s >>= 1) { if (t < s) s_red[t] += s_red[t + s]; __syncthreads(); }
    float inv = rsqrtf(s_red[0] / HH + BN_EPS); __syncthreads();

    for (int j = t; j < HH; j += 256)
        s_r[j] = silu_f((s_r[j] - mu) * inv * lng[j] + lnb[j]);
    __syncthreads();

    if (t < HD2) {
        float acc = hb1[t];
        const float* wr = hW1 + (size_t)t * HH;
        for (int j = 0; j < HH; ++j) acc += s_r[j] * wr[j];
        s_y[t] = silu_f(acc);
    }
    __syncthreads();
    part = (t < HD2) ? s_y[t] * hW2[t] : 0.f;
    s_red[t] = part; __syncthreads();
    for (int s = 128; s > 0; s >>= 1) { if (t < s) s_red[t] += s_red[t + s]; __syncthreads(); }
    if (t == 0) out[g] = s_red[0] + hb2[0];
}

// ---------------------------------------------------------------- host driver
extern "C" void kernel_launch(void* const* d_in, const int* in_sizes, int n_in,
                              void* d_out, int out_size, void* d_ws, size_t ws_size,
                              hipStream_t stream) {
    (void)in_sizes; (void)n_in; (void)out_size; (void)ws_size;
    const int*   x         = (const int*)d_in[0];
    const int*   edge_attr = (const int*)d_in[1];
    const int*   eidx      = (const int*)d_in[2];
    const int*   batch     = (const int*)d_in[3];
    const float* atom_emb  = (const float*)d_in[5];
    const float* bond_emb  = (const float*)d_in[6];
    const float* We   = (const float*)d_in[7];
    const float* be   = (const float*)d_in[8];
    const float* W1   = (const float*)d_in[9];
    const float* b1   = (const float*)d_in[10];
    const float* bn1  = (const float*)d_in[11];
    const float* W2   = (const float*)d_in[12];
    const float* b2   = (const float*)d_in[13];
    const float* bn2  = (const float*)d_in[14];
    const float* epsA = (const float*)d_in[15];
    const float* Wp   = (const float*)d_in[16];
    const float* bp   = (const float*)d_in[17];
    const float* vnW1 = (const float*)d_in[18];
    const float* vnb1 = (const float*)d_in[19];
    const float* vnbn = (const float*)d_in[20];
    const float* vnW2 = (const float*)d_in[21];
    const float* vnb2 = (const float*)d_in[22];
    const float* lng  = (const float*)d_in[23];
    const float* lnb  = (const float*)d_in[24];
    const float* hW1  = (const float*)d_in[25];
    const float* hb1  = (const float*)d_in[26];
    const float* hW2  = (const float*)d_in[27];
    const float* hb2  = (const float*)d_in[28];

    size_t off = 0;
    auto alloc = [&](size_t bytes) -> void* {
        void* p = (char*)d_ws + off;
        off += (bytes + 255) & ~(size_t)255;
        return p;
    };
    float* h   = (float*)alloc((size_t)NN * HH * 4);
    float* agg = (float*)alloc((size_t)NN * HH * 4);
    float* t1  = (float*)alloc((size_t)NN * H2 * 4);
    float* t2  = (float*)alloc((size_t)NN * HH * 4);
    float* ep  = (float*)alloc((size_t)EE * HH * 4);
    float* hp  = (float*)alloc((size_t)GG * HH * 4);
    float* hg  = (float*)alloc((size_t)GG * HH * 4);
    float* vn  = (float*)alloc((size_t)GG * HH * 4);
    float* vni = (float*)alloc((size_t)GG * HH * 4);
    float* vnt = (float*)alloc((size_t)GG * HH * 4);
    _Float16* z16    = (_Float16*)alloc((size_t)NN * HP * 2);
    _Float16* t116   = (_Float16*)alloc((size_t)NN * H2P * 2);
    _Float16* e16    = (_Float16*)alloc((size_t)EE * EDD * 2);
    _Float16* hp16   = (_Float16*)alloc((size_t)GG * HP * 2);
    _Float16* vni16  = (_Float16*)alloc((size_t)GG * HP * 2);
    _Float16* vnt16  = (_Float16*)alloc((size_t)GG * HP * 2);
    _Float16* We16   = (_Float16*)alloc((size_t)LL * HH * EDD * 2);
    _Float16* W116   = (_Float16*)alloc((size_t)LL * H2 * HP * 2);
    _Float16* W216   = (_Float16*)alloc((size_t)LL * HH * H2P * 2);
    _Float16* Wp16   = (_Float16*)alloc((size_t)LL * HH * HP * 2);
    _Float16* vnW116 = (_Float16*)alloc((size_t)HH * HP * 2);
    _Float16* vnW216 = (_Float16*)alloc((size_t)HH * HP * 2);

    const int B = 256;
    auto nb = [&](size_t n) { return (unsigned)((n + B - 1) / B); };
    auto gemm_grid = [&](int M, int Nout) { return dim3((Nout + 63) / 64, (M + 127) / 128); };

    // f16-pad weights (deterministic; cheap)
    k_cvt_pad<<<nb((size_t)LL * HH * EDD), B, 0, stream>>>(We, We16, LL * HH, EDD, EDD);
    k_cvt_pad<<<nb((size_t)LL * H2 * HP),  B, 0, stream>>>(W1, W116, LL * H2, HH, HP);
    k_cvt_pad<<<nb((size_t)LL * HH * H2P), B, 0, stream>>>(W2, W216, LL * HH, H2, H2P);
    k_cvt_pad<<<nb((size_t)LL * HH * HP),  B, 0, stream>>>(Wp, Wp16, LL * HH, HH, HP);
    k_cvt_pad<<<nb((size_t)HH * HP), B, 0, stream>>>(vnW1, vnW116, HH, HH, HP);
    k_cvt_pad<<<nb((size_t)HH * HP), B, 0, stream>>>(vnW2, vnW216, HH, HH, HP);

    // encoders
    k_gather_rows<<<nb((size_t)NN * HH), B, 0, stream>>>(atom_emb, x, h, NN, HH);
    k_gather_cvt<<<nb((size_t)EE * EDD), B, 0, stream>>>(bond_emb, edge_attr, e16, EE, EDD);

    // state init
    k_zero<<<1024, B, 0, stream>>>(hg, (size_t)GG * HH);
    k_zero<<<1024, B, 0, stream>>>(vn, (size_t)GG * HH);

    const int* srcI = eidx;
    const int* dstI = eidx + EE;

    for (int l = 0; l < LL; ++l) {
        if (l > 0) {
            // virtual node update + broadcast
            k_copy<<<1024, B, 0, stream>>>(vni, vn, (size_t)GG * HH);
            k_segment_sum<<<nb((size_t)NN * (HH / 4)), B, 0, stream>>>(h, batch, vni, NN, HH);
            k_cvt_pad<<<nb((size_t)GG * HP), B, 0, stream>>>(vni, vni16, GG, HH, HP);
            k_wmma_gemm<<<gemm_grid(GG, HH), B, 0, stream>>>(vni16, vnW116, vnb1, vnt,
                                                             GG, HH, HP, 0);
            k_bn_silu_cvt<<<nb((size_t)GG * HP), B, 0, stream>>>(vnt, vnbn, vnt16, GG, HH, HP);
            k_wmma_gemm<<<gemm_grid(GG, HH), B, 0, stream>>>(vnt16, vnW216, vnb2, vn,
                                                             GG, HH, HP, 0);
            k_add_broadcast<<<nb((size_t)NN * HH), B, 0, stream>>>(h, vn, batch);
        }
        // edge linear: ep = e_feat @ We[l].T + be[l]
        k_wmma_gemm<<<gemm_grid(EE, HH), B, 0, stream>>>(e16, We16 + (size_t)l * HH * EDD,
                                                         be + (size_t)l * HH, ep,
                                                         EE, HH, EDD, 0);
        // aggregate messages
        k_zero<<<4096, B, 0, stream>>>(agg, (size_t)NN * HH);
        k_message_scatter<<<nb((size_t)EE * (HH / 4)), B, 0, stream>>>(h, ep, srcI, dstI, agg);
        // z = (1+eps)h + agg, then MLP
        k_z_cvt<<<nb((size_t)NN * HP), B, 0, stream>>>(h, agg, epsA, l, z16);
        k_wmma_gemm<<<gemm_grid(NN, H2), B, 0, stream>>>(z16, W116 + (size_t)l * H2 * HP,
                                                         b1 + (size_t)l * H2, t1,
                                                         NN, H2, HP, 0);
        k_bn_silu_cvt<<<nb((size_t)NN * H2P), B, 0, stream>>>(t1, bn1 + (size_t)l * 4 * H2,
                                                              t116, NN, H2, H2P);
        k_wmma_gemm<<<gemm_grid(NN, HH), B, 0, stream>>>(t116, W216 + (size_t)l * HH * H2P,
                                                         b2 + (size_t)l * HH, t2,
                                                         NN, HH, H2P, 0);
        k_bn2_residual<<<nb((size_t)NN * HH), B, 0, stream>>>(t2, bn2 + (size_t)l * 4 * HH,
                                                              h, NN, HH, l > 0 ? 1 : 0);
        // pooling: hg += pool(h) @ Wp[l].T + bp[l]
        k_zero<<<1024, B, 0, stream>>>(hp, (size_t)GG * HH);
        k_segment_sum<<<nb((size_t)NN * (HH / 4)), B, 0, stream>>>(h, batch, hp, NN, HH);
        k_cvt_pad<<<nb((size_t)GG * HP), B, 0, stream>>>(hp, hp16, GG, HH, HP);
        k_wmma_gemm<<<gemm_grid(GG, HH), B, 0, stream>>>(hp16, Wp16 + (size_t)l * HH * HP,
                                                         bp + (size_t)l * HH, hg,
                                                         GG, HH, HP, 1);
    }
    // readout head
    k_readout<<<GG, B, 0, stream>>>(hg, lng, lnb, hW1, hb1, hW2, hb2, (float*)d_out);
}
